// CeptaBlock_33062658244874
// MI455X (gfx1250) — matrix operand order repaired and checked by reading, not compiled, not run on hardware
//
#include <hip/hip_runtime.h>
#include <cstdint>
#include <cstddef>

// ---------------------------------------------------------------------------
// CeptaBlock forward for MI455X (gfx1250, wave32, WMMA + async-to-LDS DMA).
// All 5 GEMMs (0.81 TFLOP) on v_wmma_f32_16x16x32_bf16, fp32 accumulate.
// Tiles staged into LDS with GLOBAL_LOAD_ASYNC_TO_LDS_B128 (ASYNCcnt,
// double-buffered, s_wait_asynccnt split-phase), fragments re-read by all
// 8 waves via ds_load_b128. 98 MB of bf16 weights resident in 192 MB L2.
// ---------------------------------------------------------------------------

typedef __bf16 bf16;
typedef __attribute__((ext_vector_type(16))) __bf16 v16bf;
typedef __attribute__((ext_vector_type(8)))  float  v8f;
typedef __attribute__((ext_vector_type(4)))  unsigned int u32x4;

union Frag { v16bf v; u32x4 q[2]; };

#define Bsz 4
#define Ssz 2048
#define Dsz 2048
#define Psz 1024
#define Hsz 7168
#define Msz (Bsz * Ssz)   /* 8192 token rows */

// LDS staging geometry: A tile 128x32 bf16, B tile 256x32 bf16, rows padded
// to 80 B (20 dwords; stride co-prime enough with 64 banks -> conflict-free
// 16-lane row-strided ds_load_b128).
#define BM 128
#define APAD 80
#define LDS_A_BYTES (BM * APAD)          /* 10240 */
#define LDS_B_BYTES (256 * APAD)         /* 20480 */
#define LDS_BUF     (LDS_A_BYTES + LDS_B_BYTES)  /* 30720; x2 = 61440 */

__device__ __forceinline__ void async_b128_to_lds(unsigned lds_addr, const void* gaddr) {
  // GLOBAL_LOAD_ASYNC_TO_LDS_B128, GV mode: per-lane LDS dest VGPR + 64-bit
  // global addr VGPR pair. Tracked by ASYNCcnt. (CDNA5 ISA)
  asm volatile("global_load_async_to_lds_b128 %0, %1, off"
               :: "v"(lds_addr), "v"((unsigned long long)(uintptr_t)gaddr)
               : "memory");
}

// ---------------- RMSNorm (fp32 in) -> bf16 out -----------------------------
__global__ __launch_bounds__(256) void rmsnorm_to_bf16(
    const float* __restrict__ x, const float* __restrict__ w,
    bf16* __restrict__ out, int D) {
  const int row = blockIdx.x;
  const float* xr = x + (size_t)row * D;
  __shared__ float red[256];
  float ss = 0.f;
  for (int i = threadIdx.x; i < D; i += 256) { float v = xr[i]; ss += v * v; }
  red[threadIdx.x] = ss; __syncthreads();
  for (int s = 128; s > 0; s >>= 1) {
    if (threadIdx.x < s) red[threadIdx.x] += red[threadIdx.x + s];
    __syncthreads();
  }
  const float rms = rsqrtf(red[0] / (float)D + 1e-6f);
  bf16* orow = out + (size_t)row * D;
  for (int i = threadIdx.x; i < D; i += 256)
    orow[i] = (bf16)(xr[i] * rms * w[i]);
}

// ---------------- fp32 -> bf16 bulk convert --------------------------------
__global__ __launch_bounds__(256) void f32_to_bf16(
    const float* __restrict__ in, bf16* __restrict__ out, long long n) {
  long long i = (long long)blockIdx.x * 256 + threadIdx.x;
  long long stride = (long long)gridDim.x * 256;
  for (; i < n; i += stride) out[i] = (bf16)in[i];
}

// ---------------- staged bf16 NT GEMM --------------------------------------
// MODE 0: C[M,N] = A*W^T + bias                      (f32 out)
// MODE 1: C[M,N] = A*W^T + bias + res                (f32 out)
// MODE 2: Y[M,H] = silu(A*Wg^T+bg) * (A*Wu^T+bu)     (bf16 out; Wg=W rows n,
//         Wu=W rows n+H; B tile = 128 gate rows + 128 up rows)
// block = 256 thr = 8 waves (2M x 4N). Wave tile 64x64 (mode 0/1) or dual
// 64x32 (mode 2); identical 4x4 WMMA loop either way.
template<int MODE>
__global__ __launch_bounds__(256) void gemm_staged(
    const bf16* __restrict__ A, const bf16* __restrict__ W,
    const float* __restrict__ bias, const float* __restrict__ res,
    float* __restrict__ Cout, bf16* __restrict__ Yout,
    int M, int N, int K, int H) {
  __shared__ char smem[2 * LDS_BUF];

  const int tid  = threadIdx.x;
  const int lane = tid & 31, wave = tid >> 5;
  const int wm = wave & 1, wn = wave >> 1;
  const int hl = lane >> 4, l16 = lane & 15;
  const int m0 = blockIdx.y * BM;
  const int nb = blockIdx.x * ((MODE == 2) ? 128 : 256);
  // generic shared-mem addr: low 32 bits are the LDS byte address (aperture)
  const unsigned ldsBase = (unsigned)(uintptr_t)smem;

  auto stage = [&](int buf, int k0) {  // 6 async b128 per thread-instruction slot
    const unsigned base = ldsBase + buf * LDS_BUF;
#pragma unroll
    for (int t = 0; t < 2; ++t) {           // A: 128 rows x 4 chunks
      const int c = tid * 2 + t, row = c >> 2, sub = c & 3;
      async_b128_to_lds(base + row * APAD + sub * 16,
                        A + (size_t)(m0 + row) * K + k0 + sub * 8);
    }
#pragma unroll
    for (int t = 0; t < 4; ++t) {           // B: 256 rows x 4 chunks
      const int c = tid * 4 + t, row = c >> 2, sub = c & 3;
      int grow;
      if (MODE == 2) grow = (row < 128) ? (nb + row) : (nb + H + (row - 128));
      else           grow = nb + row;
      async_b128_to_lds(base + LDS_A_BYTES + row * APAD + sub * 16,
                        W + (size_t)grow * K + k0 + sub * 8);
    }
  };

  const char* sm = smem;
  v8f acc[4][4] = {};
  const int KT = K / 32;

  stage(0, 0);
  for (int kt = 0; kt < KT; ++kt) {
    const int buf = kt & 1;
    if (kt + 1 < KT) {
      stage(buf ^ 1, (kt + 1) * 32);
      asm volatile("s_wait_asynccnt 0x6" ::: "memory");  // stage(kt) done
    } else {
      asm volatile("s_wait_asynccnt 0x0" ::: "memory");
    }
    __syncthreads();  // every wave's slice landed in LDS

    const char* sa = sm + buf * LDS_BUF;
    const char* sb = sa + LDS_A_BYTES;
    Frag a[4], b[4];
#pragma unroll
    for (int i = 0; i < 4; ++i) {
      const int arow = wm * 64 + i * 16 + l16;
      a[i].q[0] = *(const u32x4*)(sa + arow * APAD + hl * 16);
      a[i].q[1] = *(const u32x4*)(sa + arow * APAD + 32 + hl * 16);
    }
#pragma unroll
    for (int j = 0; j < 4; ++j) {
      int brow;
      if (MODE == 2) brow = (j < 2) ? (wn * 32 + j * 16 + l16)
                                    : (128 + wn * 32 + (j - 2) * 16 + l16);
      else           brow = wn * 64 + j * 16 + l16;
      b[j].q[0] = *(const u32x4*)(sb + brow * APAD + hl * 16);
      b[j].q[1] = *(const u32x4*)(sb + brow * APAD + 32 + hl * 16);
    }
#pragma unroll
    for (int i = 0; i < 4; ++i)
#pragma unroll
      for (int j = 0; j < 4; ++j)
        acc[i][j] = __builtin_amdgcn_wmma_f32_16x16x32_bf16(
            false, a[i].v, false, b[j].v, (short)0, acc[i][j], false, false);

    __syncthreads();  // all waves done reading buf before it is restaged
  }

  if (MODE != 2) {
#pragma unroll
    for (int i = 0; i < 4; ++i)
#pragma unroll
      for (int j = 0; j < 4; ++j) {
        const int n = nb + wn * 64 + j * 16 + l16;
        const float bi = bias[n];
#pragma unroll
        for (int r = 0; r < 8; ++r) {
          const int m = m0 + wm * 64 + i * 16 + hl * 8 + r;
          const size_t idx = (size_t)m * N + n;
          float v = acc[i][j][r] + bi;
          if (MODE == 1) v += res[idx];
          Cout[idx] = v;
        }
      }
  } else {
#pragma unroll
    for (int i = 0; i < 4; ++i)
#pragma unroll
      for (int j = 0; j < 2; ++j) {
        const int n = nb + wn * 32 + j * 16 + l16;
        const float biA = bias[n], biB = bias[n + H];
#pragma unroll
        for (int r = 0; r < 8; ++r) {
          const int m = m0 + wm * 64 + i * 16 + hl * 8 + r;
          const float av = acc[i][j][r] + biA;
          const float bv = acc[i][j + 2][r] + biB;
          const float y = (av / (1.f + __expf(-av))) * bv;  // silu(a) * b
          Yout[(size_t)m * H + n] = (bf16)y;
        }
      }
  }
}

// ---------------- hard top-alpha magnitude gate -----------------------------
__global__ __launch_bounds__(256) void topk_gate(
    float* __restrict__ u, bf16* __restrict__ t16,
    const int* __restrict__ alpha_p, int P) {
  const int row = blockIdx.x;
  float* ur = u + (size_t)row * P;
  const int alpha = *alpha_p;
  __shared__ float av[Psz];
  __shared__ int cnt;
  for (int i = threadIdx.x; i < P; i += 256) av[i] = fabsf(ur[i]);
  __syncthreads();
  unsigned thr = 0u;
  for (int bit = 30; bit >= 0; --bit) {
    const unsigned cand = thr | (1u << bit);
    if (threadIdx.x == 0) cnt = 0;
    __syncthreads();
    int c = 0;
    for (int i = threadIdx.x; i < P; i += 256)
      if (__float_as_uint(av[i]) >= cand) ++c;
    atomicAdd(&cnt, c);
    __syncthreads();
    if (cnt >= alpha) thr = cand;
    __syncthreads();
  }
  const float kth = __uint_as_float(thr);
  for (int i = threadIdx.x; i < P; i += 256) {
    const float v = ur[i];
    const float t = (fabsf(v) >= kth) ? v : 0.f;
    ur[i] = t;
    t16[(size_t)row * P + i] = (bf16)t;
  }
}

// ---------------- softmax(logits) * t -> routed (bf16) ----------------------
__global__ __launch_bounds__(256) void softmax_route(
    const float* __restrict__ logits, const float* __restrict__ t32,
    bf16* __restrict__ routed, int P) {
  const int row = blockIdx.x;
  const float* lr = logits + (size_t)row * P;
  __shared__ float red[256];
  float mx = -3.4e38f;
  for (int i = threadIdx.x; i < P; i += 256) mx = fmaxf(mx, lr[i]);
  red[threadIdx.x] = mx; __syncthreads();
  for (int s = 128; s > 0; s >>= 1) {
    if (threadIdx.x < s) red[threadIdx.x] = fmaxf(red[threadIdx.x], red[threadIdx.x + s]);
    __syncthreads();
  }
  mx = red[0]; __syncthreads();
  float sum = 0.f;
  for (int i = threadIdx.x; i < P; i += 256) sum += __expf(lr[i] - mx);
  red[threadIdx.x] = sum; __syncthreads();
  for (int s = 128; s > 0; s >>= 1) {
    if (threadIdx.x < s) red[threadIdx.x] += red[threadIdx.x + s];
    __syncthreads();
  }
  const float inv = 1.f / red[0];
  for (int i = threadIdx.x; i < P; i += 256) {
    const float g = __expf(lr[i] - mx) * inv;
    routed[(size_t)row * P + i] = (bf16)(g * t32[(size_t)row * P + i]);
  }
}

// ---------------------------------------------------------------------------
extern "C" void kernel_launch(void* const* d_in, const int* in_sizes, int n_in,
                              void* d_out, int out_size, void* d_ws, size_t ws_size,
                              hipStream_t stream) {
  (void)in_sizes; (void)n_in; (void)out_size; (void)ws_size;
  const float* x       = (const float*)d_in[0];
  const float* rms1_w  = (const float*)d_in[1];
  const float* toP_W   = (const float*)d_in[2];
  const float* toP_b   = (const float*)d_in[3];
  const float* route_W = (const float*)d_in[4];
  const float* route_b = (const float*)d_in[5];
  const float* fromP_W = (const float*)d_in[6];
  const float* fromP_b = (const float*)d_in[7];
  const float* rms2_w  = (const float*)d_in[8];
  const float* w12_W   = (const float*)d_in[9];
  const float* w12_b   = (const float*)d_in[10];
  const float* w3_W    = (const float*)d_in[11];
  const float* w3_b    = (const float*)d_in[12];
  const int*   alpha   = (const int*)d_in[13];
  float* out = (float*)d_out;

  // workspace layout (~367 MB), 256B aligned slices
  char* ws = (char*)d_ws;
  size_t off = 0;
  auto take = [&](size_t bytes) -> char* {
    char* p = ws + off; off += (bytes + 255) & ~(size_t)255; return p;
  };
  bf16* wToP16   = (bf16*)take((size_t)Psz * Dsz * 2);
  bf16* wRoute16 = (bf16*)take((size_t)Psz * Psz * 2);
  bf16* wFromP16 = (bf16*)take((size_t)Dsz * Psz * 2);
  bf16* wW12_16  = (bf16*)take((size_t)2 * Hsz * Dsz * 2);
  bf16* wW3_16   = (bf16*)take((size_t)Dsz * Hsz * 2);
  char* h1_logits = take((size_t)Msz * Dsz * 2);        // h1(bf16) / logits(f32)
  char* u_h2      = take((size_t)Msz * Psz * 4);        // u,t(f32) / h2(bf16)
  bf16* t_routed16 = (bf16*)take((size_t)Msz * Psz * 2);// t16 / routed16
  float* xres32  = (float*)take((size_t)Msz * Dsz * 4);
  bf16* y16      = (bf16*)take((size_t)Msz * Hsz * 2);

  bf16*  h1_16    = (bf16*)h1_logits;
  float* logits32 = (float*)h1_logits;
  float* u32      = (float*)u_h2;
  bf16*  h2_16    = (bf16*)u_h2;

  const dim3 blk(256);

  f32_to_bf16<<<4096, blk, 0, stream>>>(toP_W,   wToP16,   (long long)Psz * Dsz);
  f32_to_bf16<<<4096, blk, 0, stream>>>(route_W, wRoute16, (long long)Psz * Psz);
  f32_to_bf16<<<4096, blk, 0, stream>>>(fromP_W, wFromP16, (long long)Dsz * Psz);
  f32_to_bf16<<<8192, blk, 0, stream>>>(w12_W,   wW12_16,  (long long)2 * Hsz * Dsz);
  f32_to_bf16<<<8192, blk, 0, stream>>>(w3_W,    wW3_16,   (long long)Dsz * Hsz);

  // 1) h1 = rmsnorm(x)
  rmsnorm_to_bf16<<<Msz, blk, 0, stream>>>(x, rms1_w, h1_16, Dsz);
  // 2) u = h1 @ toP_W^T + b
  gemm_staged<0><<<dim3(Psz / 256, Msz / 128), blk, 0, stream>>>(
      h1_16, wToP16, toP_b, nullptr, u32, nullptr, Msz, Psz, Dsz, 0);
  // 3) hard top-alpha gate
  topk_gate<<<Msz, blk, 0, stream>>>(u32, t_routed16, alpha, Psz);
  // 4) logits = t @ route_W^T + b
  gemm_staged<0><<<dim3(Psz / 256, Msz / 128), blk, 0, stream>>>(
      t_routed16, wRoute16, route_b, nullptr, logits32, nullptr, Msz, Psz, Psz, 0);
  // 5) routed = softmax(logits) * t
  softmax_route<<<Msz, blk, 0, stream>>>(logits32, u32, t_routed16, Psz);
  // 6) xres = x + routed @ fromP_W^T + b
  gemm_staged<1><<<dim3(Dsz / 256, Msz / 128), blk, 0, stream>>>(
      t_routed16, wFromP16, fromP_b, x, xres32, nullptr, Msz, Dsz, Psz, 0);
  // 7) h2 = rmsnorm(xres)
  rmsnorm_to_bf16<<<Msz, blk, 0, stream>>>(xres32, rms2_w, h2_16, Dsz);
  // 8) y = silu(h2@W1^T+b1) * (h2@W2^T+b2)   (fused dual GEMM, bf16 out)
  gemm_staged<2><<<dim3(Hsz / 128, Msz / 128), blk, 0, stream>>>(
      h2_16, wW12_16, w12_b, nullptr, nullptr, y16, Msz, Hsz, Dsz, Hsz);
  // 9) out = xres + y @ w3_W^T + b
  gemm_staged<1><<<dim3(Dsz / 256, Msz / 128), blk, 0, stream>>>(
      y16, wW3_16, w3_b, xres32, out, nullptr, Msz, Dsz, Hsz, 0);
}
